// Attention_88441966559243
// MI455X (gfx1250) — matrix operand order repaired and checked by reading, not compiled, not run on hardware
//
#include <hip/hip_runtime.h>

typedef __attribute__((ext_vector_type(16))) __bf16 v16bf;
typedef __attribute__((ext_vector_type(8)))  float  v8f;

// ---------------------------------------------------------------------------
// Fragment loaders (wave32): A and B share one contiguous K order, not the ISA table order.
// A 16x32 bf16 : lane L holds row M=L&15; vector elem e -> K = 16*half + e
//                (contiguous 16 bf16, half = L>>4)
// B 32x16 bf16 : stored K-major (Bt[N][K]); lane L holds col N=L&15; vector
//                elem e -> K = 16*half + e  (contiguous 16 bf16)
// C/D 16x16 f32: vgpr r, lanes 0-15 -> M=r, lanes 16-31 -> M=r+8; N = L&15
// ---------------------------------------------------------------------------
static __device__ __forceinline__ v16bf a_frag(const __bf16* chunk, int half) {
  v16bf r;
#pragma unroll
  for (int i = 0; i < 8; ++i) {
    r[i]     = chunk[16 * half + i];
    r[i + 8] = chunk[16 * half + 8 + i];
  }
  return r;
}

static __device__ __forceinline__ v16bf b_frag(const __bf16* chunk, int half) {
  v16bf r;
#pragma unroll
  for (int i = 0; i < 16; ++i) r[i] = chunk[16 * half + i];
  return r;
}

static __device__ __forceinline__ v8f wmma_bf16(v16bf a, v16bf b, v8f c) {
  return __builtin_amdgcn_wmma_f32_16x16x32_bf16(
      /*neg_a=*/false, a, /*neg_b=*/false, b,
      /*c_mod=*/(short)0, c, /*reuse_a=*/false, /*reuse_b=*/false);
}

// Low 32 bits of a flat pointer into LDS == LDS byte offset (ISA 10.2).
static __device__ __forceinline__ unsigned lds_off(const void* p) {
  return (unsigned)(size_t)p;
}

// Async global->LDS copy of 64 bytes per lane (4 x B128), ASYNCcnt-tracked.
// INST_OFFSET is added to BOTH the global and the LDS address (ISA 10.7/15.18).
static __device__ __forceinline__ void async_copy64(unsigned lds_dst,
                                                    const void* gsrc) {
  asm volatile(
      "global_load_async_to_lds_b128 %0, %1, off\n\t"
      "global_load_async_to_lds_b128 %0, %1, off offset:16\n\t"
      "global_load_async_to_lds_b128 %0, %1, off offset:32\n\t"
      "global_load_async_to_lds_b128 %0, %1, off offset:48" ::"v"(lds_dst),
      "v"(gsrc)
      : "memory");
}

// Async global->LDS copy of 32 bytes per lane (2 x B128).
static __device__ __forceinline__ void async_copy32(unsigned lds_dst,
                                                    const void* gsrc) {
  asm volatile(
      "global_load_async_to_lds_b128 %0, %1, off\n\t"
      "global_load_async_to_lds_b128 %0, %1, off offset:16" ::"v"(lds_dst),
      "v"(gsrc)
      : "memory");
}

static __device__ __forceinline__ void wait_async0() {
  asm volatile("s_wait_asynccnt 0" ::: "memory");
}

// ---------------------------------------------------------------------------
// Kernel 1: projection GEMM.  dst[(b,h,n,d)] (bf16) = scale * (x @ W[:,col0:])
// x: [4096,1024] f32 row-major.  W: [1024, ldw] f32, take 1024 cols at col0.
// Block = 128 threads (4 waves), tile 64(M) x 64(N), K-step 32.
// (fp32 -> bf16 conversion forces the VGPR staging path here.)
// ---------------------------------------------------------------------------
__global__ __launch_bounds__(128) void proj_kernel(
    const float* __restrict__ x, const float* __restrict__ W,
    int ldw, int col0, float scale, __bf16* __restrict__ dst) {
  const int tid  = threadIdx.x;
  const int wave = tid >> 5, lane = tid & 31, half = lane >> 4, lr = lane & 15;
  const int m0 = blockIdx.x * 64;
  const int n0 = blockIdx.y * 64;

  __shared__ __attribute__((aligned(16))) __bf16 sA[64][32];  // [m][k]
  __shared__ __attribute__((aligned(16))) __bf16 sB[64][32];  // [n][k]

  v8f acc[4];
#pragma unroll
  for (int c = 0; c < 4; ++c)
#pragma unroll
    for (int i = 0; i < 8; ++i) acc[c][i] = 0.f;

  const int trow = tid >> 1;        // 0..63
  const int tc   = (tid & 1) * 16;  // 0 or 16

  for (int k0 = 0; k0 < 1024; k0 += 32) {
    const float* xr = x + (size_t)(m0 + trow) * 1024 + k0 + tc;
#pragma unroll
    for (int i = 0; i < 16; ++i) sA[trow][tc + i] = (__bf16)xr[i];
#pragma unroll
    for (int i = 0; i < 16; ++i)
      sB[trow][tc + i] =
          (__bf16)W[(size_t)(k0 + tc + i) * ldw + col0 + n0 + trow];
    __syncthreads();

    v16bf a = a_frag(&sA[wave * 16 + lr][0], half);
#pragma unroll
    for (int c = 0; c < 4; ++c) {
      v16bf b = b_frag(&sB[c * 16 + lr][0], half);
      acc[c] = wmma_bf16(a, b, acc[c]);
    }
    __syncthreads();
  }

#pragma unroll
  for (int c = 0; c < 4; ++c)
#pragma unroll
    for (int r = 0; r < 8; ++r) {
      const int gm = m0 + wave * 16 + r + 8 * half;  // 0..4095
      const int gc = n0 + c * 16 + lr;               // 0..1023
      const int bb = gm >> 11, nn = gm & 2047;
      const int hh = gc >> 6, dd = gc & 63;
      dst[(((size_t)bb * 16 + hh) * 2048 + nn) * 64 + dd] =
          (__bf16)(acc[c][r] * scale);
    }
}

// ---------------------------------------------------------------------------
// Kernel 2: flash attention per (b, h, 64-row q tile).  128 threads, 4 waves,
// each wave owns 16 query rows.  Key tiles of 64, online softmax in fp32.
// Q/K tiles staged with GLOBAL_LOAD_ASYNC_TO_LDS (no VGPR round trip);
// V needs a transpose so it stays on the VGPR path.
// ---------------------------------------------------------------------------
__global__ __launch_bounds__(128) void attn_kernel(
    const __bf16* __restrict__ qg, const __bf16* __restrict__ kg,
    const __bf16* __restrict__ vg, const float* __restrict__ bias,
    const int* __restrict__ mask, __bf16* __restrict__ attn_out) {
  const int tid  = threadIdx.x;
  const int wave = tid >> 5, lane = tid & 31, half = lane >> 4, lr = lane & 15;

  int idx = blockIdx.x;
  const int qt = idx & 31; idx >>= 5;
  const int h  = idx & 15; idx >>= 4;
  const int b  = idx;           // 0..1
  const int n0 = qt * 64;

  const __bf16* Q = qg + ((size_t)b * 16 + h) * 2048 * 64;
  const __bf16* K = kg + ((size_t)b * 16 + h) * 2048 * 64;
  const __bf16* V = vg + ((size_t)b * 16 + h) * 2048 * 64;
  const float* Bb = bias + (size_t)b * 2048 * 2048;
  const int*   Mb = mask + (size_t)b * 2048;

  __shared__ __attribute__((aligned(16))) __bf16 sQ[64][64];     // [qrow][d]
  __shared__ __attribute__((aligned(16))) __bf16 sK[64][64];     // [key][d]
  __shared__ __attribute__((aligned(16))) __bf16 sVt[64][64];    // [d][key]
  __shared__ __attribute__((aligned(16))) __bf16 sP[4][16][64];  // [row][key]

  const int trow = tid >> 1;        // 0..63
  const int tc   = (tid & 1) * 32;  // 0 or 32

  // ---- stage Q tile via async copy (bf16, layout-preserving) ----
  async_copy64(lds_off(&sQ[trow][tc]), Q + (size_t)(n0 + trow) * 64 + tc);
  wait_async0();
  __syncthreads();
  const v16bf aQ0 = a_frag(&sQ[wave * 16 + lr][0], half);
  const v16bf aQ1 = a_frag(&sQ[wave * 16 + lr][32], half);

  float m_row[8], l_row[8];
  v8f o[4];
#pragma unroll
  for (int c = 0; c < 4; ++c)
#pragma unroll
    for (int i = 0; i < 8; ++i) o[c][i] = 0.f;
#pragma unroll
  for (int r = 0; r < 8; ++r) { m_row[r] = -3.4e38f; l_row[r] = 0.f; }

  for (int kt = 0; kt < 32; ++kt) {
    const int kbase = kt * 64;

    // ---- stage K tile via async copy; V tile transposed via VGPRs ----
    async_copy64(lds_off(&sK[trow][tc]), K + (size_t)(kbase + trow) * 64 + tc);
    {
      const __bf16* vs = V + (size_t)(kbase + trow) * 64 + tc;
#pragma unroll
      for (int i = 0; i < 32; ++i) sVt[tc + i][trow] = vs[i];
    }
    // prefetch next tile's bias rows into cache while we compute this one
    if (kbase + 64 < 2048)
      __builtin_prefetch(
          Bb + ((size_t)(n0 + wave * 16 + lr) * 2048 + kbase + 64), 0, 3);
    wait_async0();
    __syncthreads();

    // S = Q K^T  (q pre-scaled by DIM_HEAD^-0.5 in projection)
    v8f s[4];
#pragma unroll
    for (int c = 0; c < 4; ++c) {
#pragma unroll
      for (int i = 0; i < 8; ++i) s[c][i] = 0.f;
      v16bf b0 = b_frag(&sK[c * 16 + lr][0], half);
      s[c] = wmma_bf16(aQ0, b0, s[c]);
      v16bf b1 = b_frag(&sK[c * 16 + lr][32], half);
      s[c] = wmma_bf16(aQ1, b1, s[c]);
    }

    // bias + key mask + online softmax (row M = r + 8*half, col N = lr)
    float p[4][8];
#pragma unroll
    for (int r = 0; r < 8; ++r) {
      const int gq = n0 + wave * 16 + r + 8 * half;
      float mx = -3.4e38f;
#pragma unroll
      for (int c = 0; c < 4; ++c) {
        const int gk = kbase + c * 16 + lr;
        float v = s[c][r] + Bb[(size_t)gq * 2048 + gk];
        if (Mb[gk] == 0) v = -3.4e38f;
        p[c][r] = v;
        mx = fmaxf(mx, v);
      }
#pragma unroll
      for (int off = 1; off < 16; off <<= 1)
        mx = fmaxf(mx, __shfl_xor(mx, off, 32));

      const float mnew  = fmaxf(m_row[r], mx);
      const float alpha = __expf(m_row[r] - mnew);
      float rs = 0.f;
#pragma unroll
      for (int c = 0; c < 4; ++c) {
        const float e = __expf(p[c][r] - mnew);
        p[c][r] = e;
        rs += e;
      }
#pragma unroll
      for (int off = 1; off < 16; off <<= 1) rs += __shfl_xor(rs, off, 32);

      l_row[r] = l_row[r] * alpha + rs;
      m_row[r] = mnew;
#pragma unroll
      for (int c = 0; c < 4; ++c) o[c][r] *= alpha;
#pragma unroll
      for (int c = 0; c < 4; ++c)
        sP[wave][r + 8 * half][c * 16 + lr] = (__bf16)p[c][r];
    }

    // in-wave LDS exchange (C-layout -> A-layout): wait on DS counter
    asm volatile("s_wait_dscnt 0" ::: "memory");

    // O += P V
#pragma unroll
    for (int kk = 0; kk < 2; ++kk) {
      v16bf aP = a_frag(&sP[wave][lr][kk * 32], half);
#pragma unroll
      for (int c = 0; c < 4; ++c) {
        v16bf bv = b_frag(&sVt[c * 16 + lr][kk * 32], half);
        o[c] = wmma_bf16(aP, bv, o[c]);
      }
    }
    __syncthreads();
  }

  // normalize and emit merged-heads row-major [B*N, H*64] in bf16
#pragma unroll
  for (int r = 0; r < 8; ++r) {
    const float inv = 1.f / l_row[r];
    const int gn = n0 + wave * 16 + r + 8 * half;
    const size_t rowoff = ((size_t)b * 2048 + gn) * 1024 + (size_t)h * 64;
#pragma unroll
    for (int c = 0; c < 4; ++c)
      attn_out[rowoff + c * 16 + lr] = (__bf16)(o[c][r] * inv);
  }
}

// ---------------------------------------------------------------------------
// Kernel 3: output projection.  out[4096,1024] f32 = A(bf16) @ Wo + bo
// A tile staged via async copy (bf16, layout-preserving); Wo converts f32->bf16
// so it stays on the VGPR path.
// ---------------------------------------------------------------------------
__global__ __launch_bounds__(128) void oproj_kernel(
    const __bf16* __restrict__ A, const float* __restrict__ W,
    const float* __restrict__ bo, float* __restrict__ out) {
  const int tid  = threadIdx.x;
  const int wave = tid >> 5, lane = tid & 31, half = lane >> 4, lr = lane & 15;
  const int m0 = blockIdx.x * 64;
  const int n0 = blockIdx.y * 64;

  __shared__ __attribute__((aligned(16))) __bf16 sA[64][32];
  __shared__ __attribute__((aligned(16))) __bf16 sB[64][32];

  v8f acc[4];
#pragma unroll
  for (int c = 0; c < 4; ++c)
#pragma unroll
    for (int i = 0; i < 8; ++i) acc[c][i] = 0.f;

  const int trow = tid >> 1;
  const int tc   = (tid & 1) * 16;

  for (int k0 = 0; k0 < 1024; k0 += 32) {
    async_copy32(lds_off(&sA[trow][tc]),
                 A + (size_t)(m0 + trow) * 1024 + k0 + tc);
#pragma unroll
    for (int i = 0; i < 16; ++i)
      sB[trow][tc + i] = (__bf16)W[(size_t)(k0 + tc + i) * 1024 + n0 + trow];
    wait_async0();
    __syncthreads();

    v16bf a = a_frag(&sA[wave * 16 + lr][0], half);
#pragma unroll
    for (int c = 0; c < 4; ++c) {
      v16bf b = b_frag(&sB[c * 16 + lr][0], half);
      acc[c] = wmma_bf16(a, b, acc[c]);
    }
    __syncthreads();
  }

#pragma unroll
  for (int c = 0; c < 4; ++c)
#pragma unroll
    for (int r = 0; r < 8; ++r) {
      const int gm = m0 + wave * 16 + r + 8 * half;
      const int gc = n0 + c * 16 + lr;
      out[(size_t)gm * 1024 + gc] = acc[c][r] + bo[gc];
    }
}

// ---------------------------------------------------------------------------
// host side
// ---------------------------------------------------------------------------
extern "C" void kernel_launch(void* const* d_in, const int* in_sizes, int n_in,
                              void* d_out, int out_size, void* d_ws,
                              size_t ws_size, hipStream_t stream) {
  const float* x    = (const float*)d_in[0];  // [2,2048,1024]
  const float* bias = (const float*)d_in[1];  // [2,2048,2048]
  const int*   mask = (const int*)d_in[2];    // [2,2048]
  const float* Wq   = (const float*)d_in[3];  // [1024,1024]
  const float* Wkv  = (const float*)d_in[4];  // [1024,2048]
  const float* Wo   = (const float*)d_in[5];  // [1024,1024]
  const float* bo   = (const float*)d_in[6];  // [1024]
  float* out = (float*)d_out;

  char* ws = (char*)d_ws;
  const size_t SZ = (size_t)8 * 1024 * 1024;  // 4M bf16 elems each
  __bf16* qws = (__bf16*)(ws + 0 * SZ);       // [2,16,2048,64]
  __bf16* kws = (__bf16*)(ws + 1 * SZ);
  __bf16* vws = (__bf16*)(ws + 2 * SZ);
  __bf16* aws = (__bf16*)(ws + 3 * SZ);       // [2*2048, 1024]

  dim3 blk(128);
  dim3 gProj(64, 16);  // 4096/64 x 1024/64

  // q = (x @ Wq) * DIM_HEAD^-0.5 ; k,v from Wkv halves
  proj_kernel<<<gProj, blk, 0, stream>>>(x, Wq, 1024, 0, 0.125f, qws);
  proj_kernel<<<gProj, blk, 0, stream>>>(x, Wkv, 2048, 0, 1.0f, kws);
  proj_kernel<<<gProj, blk, 0, stream>>>(x, Wkv, 2048, 1024, 1.0f, vws);

  attn_kernel<<<dim3(2 * 16 * 32), blk, 0, stream>>>(qws, kws, vws, bias, mask,
                                                     aws);

  oproj_kernel<<<gProj, blk, 0, stream>>>(aws, Wo, bo, out);
}
